// GNNFiLMLayer_35158602285600
// MI455X (gfx1250) — matrix operand order repaired and checked by reading, not compiled, not run on hardware
//
#include <hip/hip_runtime.h>

typedef __attribute__((ext_vector_type(16))) _Float16 v16h;
typedef __attribute__((ext_vector_type(8)))  _Float16 v8h;
typedef __attribute__((ext_vector_type(8)))  float    v8f;
typedef __attribute__((ext_vector_type(4)))  int      v4i;

#define DD 128
#define TILE_M 128
#define LDS_STRIDE 136   // halves; 272B row stride = 68 dwords -> 4-bank rotation

#if defined(__has_builtin)
#if __has_builtin(__builtin_amdgcn_global_load_async_to_lds_b128)
#define HAVE_ASYNC_LDS 1
#endif
#endif

#ifdef HAVE_ASYNC_LDS
__device__ __forceinline__ void async_cp16(const _Float16* g, _Float16* l) {
    __builtin_amdgcn_global_load_async_to_lds_b128((v4i*)g, (v4i*)l, 0, 0);
}
__device__ __forceinline__ void async_wait0() {
#if __has_builtin(__builtin_amdgcn_s_wait_asynccnt)
    __builtin_amdgcn_s_wait_asynccnt(0);
#else
    asm volatile("s_wait_asynccnt 0x0" ::: "memory");
#endif
}
#endif

// ---------------- zero h ----------------
__global__ __launch_bounds__(256) void zero_f4(float4* p, int n4) {
    int i = blockIdx.x * 256 + threadIdx.x;
    if (i < n4) p[i] = make_float4(0.f, 0.f, 0.f, 0.f);
}

// ---------------- feat fp32 -> f16 ----------------
__global__ __launch_bounds__(256) void cvt_feat(const float* __restrict__ f,
                                                _Float16* __restrict__ o, int n) {
    int i = blockIdx.x * 256 + threadIdx.x;
    if (i < n) o[i] = (_Float16)f[i];
}

// ---- pack [W | Wf] transposed+f16: WT[t][col(0..383)][k(0..127)] ----
__global__ __launch_bounds__(256) void pack_w(const float* __restrict__ W,
                                              const float* __restrict__ Wf,
                                              _Float16* __restrict__ WT, int T) {
    int i = blockIdx.x * 256 + threadIdx.x;
    int total = T * 384 * DD;
    if (i >= total) return;
    int k   = i & (DD - 1);
    int col = (i >> 7) % 384;
    int t   = i / (384 * DD);
    float v;
    if (col < DD) v = W [((size_t)t * DD + k) * DD  + col];
    else          v = Wf[((size_t)t * DD + k) * 256 + (col - DD)];
    WT[i] = (_Float16)v;
}

// ---------------- GEMM + FiLM epilogue ----------------
// m[n, 0:128] = relu( (feat@Wf)[:, :128] * (feat@W) + (feat@Wf)[:, 128:] )
// WT is column-major combined weights for one edge type: [col(384)][k(128)] f16.
__global__ __launch_bounds__(256) void gemm_film(const _Float16* __restrict__ featH,
                                                 const _Float16* __restrict__ WT,
                                                 float* __restrict__ m, int N) {
    __shared__ _Float16 fs [TILE_M * LDS_STRIDE];   //  34,816 B feat tile
    __shared__ _Float16 wls[384    * LDS_STRIDE];   // 104,448 B full weight panel

    const int tid     = threadIdx.x;
    const int rowBase = blockIdx.x * TILE_M;

    // ---- stage full weight panel: 384 cols x 128 halves (16B chunks) ----
    for (int c = tid; c < 384 * (DD / 8); c += 256) {
        int col = c >> 4;
        int cc  = c & 15;
        _Float16*       ldst = &wls[col * LDS_STRIDE + cc * 8];
        const _Float16* gsrc = WT + (size_t)col * DD + cc * 8;
#ifdef HAVE_ASYNC_LDS
        async_cp16(gsrc, ldst);
#else
        *(v8h*)ldst = *(const v8h*)gsrc;
#endif
    }
    // ---- stage feat tile: 128 rows x 128 halves ----
    for (int c = tid; c < TILE_M * (DD / 8); c += 256) {
        int r  = c >> 4;
        int cc = c & 15;
        int grow = rowBase + r;
        _Float16* ldst = &fs[r * LDS_STRIDE + cc * 8];
        if (grow < N) {
            const _Float16* gsrc = featH + (size_t)grow * DD + cc * 8;
#ifdef HAVE_ASYNC_LDS
            async_cp16(gsrc, ldst);
#else
            *(v8h*)ldst = *(const v8h*)gsrc;
#endif
        } else {
            v8h z = {};
            *(v8h*)ldst = z;
        }
    }
#ifdef HAVE_ASYNC_LDS
    async_wait0();
#endif
    __syncthreads();

    const int wave = tid >> 5;
    const int lane = tid & 31;
    const int lmod = lane & 15;
    const int g    = lane >> 4;       // half-wave group
    const int wrow = wave * 16;       // this wave's 16-row band
    const bool fullTile = (rowBase + TILE_M) <= N;   // uniform: 781/782 blocks

    for (int ci = 0; ci < 8; ++ci) {  // 8 column tiles of 16
        const int col = ci * 16 + lmod;
        v8f accM = {}, accG = {}, accB = {};
#pragma unroll
        for (int kc = 0; kc < 4; ++kc) {   // K chunks of 32
            const int k0 = kc * 32;
            // A fragment (16x32 f16): lane group g holds K in {16g..16g+15}
            const _Float16* arow = &fs[(wrow + lmod) * LDS_STRIDE];
            v8h a0 = *(const v8h*)(arow + k0 + 16 * g);
            v8h a1 = *(const v8h*)(arow + k0 + 16 * g + 8);
            v16h A = __builtin_shufflevector(a0, a1, 0,1,2,3,4,5,6,7,
                                                     8,9,10,11,12,13,14,15);
            // B fragments (32x16): lane = column, 16 contiguous K starting at 16g
            const _Float16* bp = &wls[col * LDS_STRIDE + k0 + 16 * g];
            v8h bm0 = *(const v8h*)bp;
            v8h bm1 = *(const v8h*)(bp + 8);
            v8h bg0 = *(const v8h*)(bp + 128 * LDS_STRIDE);
            v8h bg1 = *(const v8h*)(bp + 128 * LDS_STRIDE + 8);
            v8h bb0 = *(const v8h*)(bp + 256 * LDS_STRIDE);
            v8h bb1 = *(const v8h*)(bp + 256 * LDS_STRIDE + 8);
            v16h Bm = __builtin_shufflevector(bm0, bm1, 0,1,2,3,4,5,6,7,
                                                        8,9,10,11,12,13,14,15);
            v16h Bg = __builtin_shufflevector(bg0, bg1, 0,1,2,3,4,5,6,7,
                                                        8,9,10,11,12,13,14,15);
            v16h Bb = __builtin_shufflevector(bb0, bb1, 0,1,2,3,4,5,6,7,
                                                        8,9,10,11,12,13,14,15);
            accM = __builtin_amdgcn_wmma_f32_16x16x32_f16(false, A, false, Bm,
                                                          (short)0, accM, false, false);
            accG = __builtin_amdgcn_wmma_f32_16x16x32_f16(false, A, false, Bg,
                                                          (short)0, accG, false, false);
            accB = __builtin_amdgcn_wmma_f32_16x16x32_f16(false, A, false, Bb,
                                                          (short)0, accB, false, false);
        }
        // FiLM + ReLU on C/D layout: VGPR r, lane l -> row = r + 8*(l>>4), col = l&15
        const int rBase = rowBase + wrow + 8 * g;
        float* mp = m + (size_t)rBase * DD + ci * 16 + lmod;
        if (fullTile) {
#pragma unroll
            for (int r = 0; r < 8; ++r) {
                float v = accG[r] * accM[r] + accB[r];
                mp[(size_t)r * DD] = v > 0.f ? v : 0.f;
            }
        } else {
#pragma unroll
            for (int r = 0; r < 8; ++r) {
                float v = accG[r] * accM[r] + accB[r];
                v = v > 0.f ? v : 0.f;
                if (rBase + r < N) mp[(size_t)r * DD] = v;
            }
        }
    }
}

// ---------------- edge gather + atomic scatter-add ----------------
// one wave per edge: 128 floats = float4 per lane, 4 f32 atomics per lane
__global__ __launch_bounds__(256) void edge_scatter(const float* __restrict__ m,
                                                    const int* __restrict__ src,
                                                    const int* __restrict__ dst,
                                                    float* __restrict__ h, int E) {
    int e    = (blockIdx.x * 256 + threadIdx.x) >> 5;
    int lane = threadIdx.x & 31;
    if (e >= E) return;
    int s = src[e], d = dst[e];
    const float4 v = *(const float4*)(m + (size_t)s * DD + lane * 4);
    float* hp = h + (size_t)d * DD + lane * 4;
    atomicAdd(hp + 0, v.x);
    atomicAdd(hp + 1, v.y);
    atomicAdd(hp + 2, v.z);
    atomicAdd(hp + 3, v.w);
}

// ---------------- LayerNorm: one wave32 per row ----------------
__global__ __launch_bounds__(256) void layernorm(const float* __restrict__ h,
                                                 const float* __restrict__ lw,
                                                 const float* __restrict__ lb,
                                                 float* __restrict__ out, int N) {
    int row  = (blockIdx.x * 256 + threadIdx.x) >> 5;
    int lane = threadIdx.x & 31;
    if (row >= N) return;
    const float4 v = *(const float4*)(h + (size_t)row * DD + lane * 4);
    float s = v.x + v.y + v.z + v.w;
    float q = v.x * v.x + v.y * v.y + v.z * v.z + v.w * v.w;
#pragma unroll
    for (int off = 16; off; off >>= 1) {
        s += __shfl_xor(s, off, 32);
        q += __shfl_xor(q, off, 32);
    }
    float mu  = s * (1.f / 128.f);
    float var = q * (1.f / 128.f) - mu * mu;
    float rs  = rsqrtf(var + 1e-5f);
    const float4 w = *(const float4*)(lw + lane * 4);
    const float4 b = *(const float4*)(lb + lane * 4);
    float4 o;
    o.x = (v.x - mu) * rs * w.x + b.x;
    o.y = (v.y - mu) * rs * w.y + b.y;
    o.z = (v.z - mu) * rs * w.z + b.z;
    o.w = (v.w - mu) * rs * w.w + b.w;
    *(float4*)(out + (size_t)row * DD + lane * 4) = o;
}

extern "C" void kernel_launch(void* const* d_in, const int* in_sizes, int n_in,
                              void* d_out, int out_size, void* d_ws, size_t ws_size,
                              hipStream_t stream) {
    const float* feat = (const float*)d_in[0];
    const float* W    = (const float*)d_in[1];
    const float* Wf   = (const float*)d_in[2];
    const float* ln_w = (const float*)d_in[3];
    const float* ln_b = (const float*)d_in[4];
    const int*   src  = (const int*)d_in[5];
    const int*   dst  = (const int*)d_in[6];

    const int N = in_sizes[0] / DD;
    const int T = in_sizes[1] / (DD * DD);
    const int E = in_sizes[5] / T;

    // workspace layout (all 256B aligned)
    char* ws = (char*)d_ws;
    size_t off = 0;
    float*    h     = (float*)(ws + off);    off += (size_t)N * DD * 4;   // 51.2 MB
    _Float16* featH = (_Float16*)(ws + off); off += (size_t)N * DD * 2;   // 25.6 MB
    _Float16* WT    = (_Float16*)(ws + off); off += (size_t)T * 384 * DD * 2; // 0.4 MB
    float*    m     = (float*)(ws + off);    off += (size_t)N * DD * 4;   // 51.2 MB
    (void)ws_size;

    const int nFeat = N * DD;
    zero_f4 <<<(nFeat / 4 + 255) / 256, 256, 0, stream>>>((float4*)h, nFeat / 4);
    cvt_feat<<<(nFeat + 255) / 256,     256, 0, stream>>>(feat, featH, nFeat);
    pack_w  <<<(T * 384 * DD + 255) / 256, 256, 0, stream>>>(W, Wf, WT, T);

    const int gemmBlocks = (N + TILE_M - 1) / TILE_M;
    const int edgeBlocks = (E + 7) / 8;
    for (int t = 0; t < T; ++t) {
        gemm_film   <<<gemmBlocks, 256, 0, stream>>>(featH, WT + (size_t)t * 384 * DD, m, N);
        edge_scatter<<<edgeBlocks, 256, 0, stream>>>(m, src + (size_t)t * E,
                                                     dst + (size_t)t * E, h, E);
    }
    layernorm<<<(N * 32 + 255) / 256, 256, 0, stream>>>(h, ln_w, ln_b, (float*)d_out, N);
}
